// SwinResidualCrossAttention_34076270526731
// MI455X (gfx1250) — matrix operand order repaired and checked by reading, not compiled, not run on hardware
//
#include <hip/hip_runtime.h>

typedef __attribute__((ext_vector_type(16))) __bf16 v16bf;
typedef __attribute__((ext_vector_type(8)))  __bf16 v8bf;
typedef __attribute__((ext_vector_type(8)))  float  v8f;
typedef __attribute__((ext_vector_type(8)))  unsigned short v8us;
typedef __attribute__((ext_vector_type(4)))  unsigned short v4us;

#define WIN_TOT 648          // 8 * 9 * 9 windows
#define S_TOK   64           // tokens per window (8x8)
#define CDIM    512
#define M_TOT   (WIN_TOT * S_TOK)   // 41472 tokens
#define QK_LD   1024         // Q | K packed per token (V goes to transposed buffer)

// ---------- bf16 helpers (round-to-nearest-even) ----------
__device__ inline unsigned short f2bf_bits(float f) {
    union { float f; unsigned u; } v; v.f = f;
    unsigned u = v.u;
    if ((u & 0x7fffffffu) > 0x7f800000u) return (unsigned short)((u >> 16) | 0x0040u); // NaN
    unsigned r = u + 0x7fffu + ((u >> 16) & 1u);
    return (unsigned short)(r >> 16);
}

// A-fragment (16x32 bf16): lane halves j=0..7 -> K = col0 + j, j=8..15 -> K = col0 + 16 + (j-8)
// => two contiguous 16B runs from a row-major K tile.
__device__ inline v16bf frag_split16(const unsigned short* rowp, int col0) {
    v8bf lo = *reinterpret_cast<const v8bf*>(rowp + col0);
    v8bf hi = *reinterpret_cast<const v8bf*>(rowp + col0 + 16);
    return __builtin_shufflevector(lo, hi, 0,1,2,3,4,5,6,7,8,9,10,11,12,13,14,15);
}
// B-fragment (32x16 bf16): lane halves j=0..15 -> K = base + j at fixed column
// => one contiguous 32B run from an N-major tile.
__device__ inline v16bf frag_contig(const unsigned short* p) {
    v8bf lo = *reinterpret_cast<const v8bf*>(p);
    v8bf hi = *reinterpret_cast<const v8bf*>(p + 8);
    return __builtin_shufflevector(lo, hi, 0,1,2,3,4,5,6,7,8,9,10,11,12,13,14,15);
}

// ---------- kernel 1: gather padded windows, convert to bf16 (8 channels/thread) ----------
__global__ __launch_bounds__(256)
void window_bf16(const float* __restrict__ src, unsigned short* __restrict__ dst) {
    int idx = blockIdx.x * 256 + threadIdx.x;            // one 8-channel chunk
    const int total8 = WIN_TOT * S_TOK * CDIM / 8;
    if (idx >= total8) return;
    int c8   = (idx & 63) << 3;
    int rest = idx >> 6;
    int s    = rest & 63;
    int win  = rest >> 6;
    int b    = win / 81;
    int r81  = win - b * 81;
    int hy   = r81 / 9;
    int wx   = r81 - hy * 9;
    int hp   = hy * 8 + (s >> 3) - 4;
    int wp   = wx * 8 + (s & 7) - 4;
    v8us o = (v8us)(unsigned short)0;
    if (hp >= 0 && hp < 64 && wp >= 0 && wp < 64) {
        const float* p = src + (((size_t)b * 64 + hp) * 64 + wp) * CDIM + c8;
        float4 a = *(const float4*)p;
        float4 c = *(const float4*)(p + 4);
        o[0] = f2bf_bits(a.x); o[1] = f2bf_bits(a.y); o[2] = f2bf_bits(a.z); o[3] = f2bf_bits(a.w);
        o[4] = f2bf_bits(c.x); o[5] = f2bf_bits(c.y); o[6] = f2bf_bits(c.z); o[7] = f2bf_bits(c.w);
    }
    *(v8us*)&dst[(size_t)idx * 8] = o;
}

// ---------- kernel 2: f32 -> bf16 flat copy (weights), 8 elems/thread ----------
__global__ __launch_bounds__(256)
void cvt_bf16(const float* __restrict__ s, unsigned short* __restrict__ d, int n8) {
    int i = blockIdx.x * 256 + threadIdx.x;
    if (i >= n8) return;
    const float* p = s + (size_t)i * 8;
    float4 a = *(const float4*)p;
    float4 c = *(const float4*)(p + 4);
    v8us o;
    o[0] = f2bf_bits(a.x); o[1] = f2bf_bits(a.y); o[2] = f2bf_bits(a.z); o[3] = f2bf_bits(a.w);
    o[4] = f2bf_bits(c.x); o[5] = f2bf_bits(c.y); o[6] = f2bf_bits(c.z); o[7] = f2bf_bits(c.w);
    *(v8us*)&d[(size_t)i * 8] = o;
}

// ---------- kernels 3/5: tiled bf16 WMMA GEMM  O[m][n] = A[m][:] . W[n][:] + bias[n] ----------
// BM=64, BN=64, BK=32; 128 threads = 4 waves; wave owns a 16-row strip x 64 cols.
// mode 0 (QKV proj): N in [0,512) uses Aq, else Akv; Q/K -> QK[tok][1024] bf16,
//                    V (n>=1024) -> Vt[win][c][s] bf16 (transposed for attention B-operand).
// mode 1 (out proj): f32 -> Of[m][512].
__global__ __launch_bounds__(128)
void gemm_wmma_bf16(const unsigned short* __restrict__ Aq,
                    const unsigned short* __restrict__ Akv,
                    const unsigned short* __restrict__ Bw,   // [Ntot][K] bf16 row-major
                    const float* __restrict__ bias,
                    unsigned short* __restrict__ QK,
                    unsigned short* __restrict__ Vt,
                    float* __restrict__ Of,
                    int K, int mode) {
    __shared__ __attribute__((aligned(16))) unsigned short As [64][40];  // row-major K tile
    __shared__ __attribute__((aligned(16))) unsigned short BsT[64][40];  // N-major tile (natural W layout)
    const int tid  = threadIdx.x;
    const int m0   = blockIdx.x * 64;
    const int n0   = blockIdx.y * 64;
    const unsigned short* __restrict__ A = (mode == 0 && n0 >= CDIM) ? Akv : Aq;
    const int wv   = tid >> 5;
    const int lane = tid & 31;
    const int lrow = lane & 15;
    const int lhi  = lane >> 4;

    v8f acc[4];
    #pragma unroll
    for (int s = 0; s < 4; ++s) acc[s] = (v8f){0.f,0.f,0.f,0.f,0.f,0.f,0.f,0.f};

    for (int k0 = 0; k0 < K; k0 += 32) {
        // stage tiles: 64 rows x 32 halves each = 256 16-byte chunks; 2 per thread per tile
        #pragma unroll
        for (int i = 0; i < 2; ++i) {
            int idx = tid + i * 128;
            int r = idx >> 2, c8 = (idx & 3) << 3;
            *(v8us*)&As [r][c8] = *(const v8us*)&A [(size_t)(m0 + r) * K + k0 + c8];
            *(v8us*)&BsT[r][c8] = *(const v8us*)&Bw[(size_t)(n0 + r) * K + k0 + c8];
        }
        // prefetch next K tile while this one is consumed
        if (k0 + 32 < K) {
            int r = tid >> 2, c8 = (tid & 3) << 3;
            __builtin_prefetch(&A [(size_t)(m0 + r) * K + k0 + 32 + c8], 0, 3);
            __builtin_prefetch(&Bw[(size_t)(n0 + r) * K + k0 + 32 + c8], 0, 3);
        }
        __syncthreads();

        v16bf a = frag_split16(&As[wv * 16 + lrow][0], lhi << 3);
        #pragma unroll
        for (int s = 0; s < 4; ++s) {
            v16bf bb = frag_split16(&BsT[s * 16 + lrow][0], lhi << 3);
            acc[s] = __builtin_amdgcn_wmma_f32_16x16x32_bf16(
                         false, a, false, bb, (short)0, acc[s], false, false);
        }
        __syncthreads();
    }

    // C/D layout: acc[r] -> (M = r + 8*lhi, N = lane%16)
    #pragma unroll
    for (int s = 0; s < 4; ++s) {
        #pragma unroll
        for (int r = 0; r < 8; ++r) {
            int mrow = m0 + wv * 16 + (lhi << 3) + r;
            int ncol = n0 + s * 16 + lrow;
            float v = acc[s][r] + bias[ncol];
            if (mode == 0) {
                if (ncol < 1024)
                    QK[(size_t)mrow * QK_LD + ncol] = f2bf_bits(v);
                else   // V: store transposed per window: Vt[win][channel][token-in-window]
                    Vt[(((size_t)(mrow >> 6)) * CDIM + (ncol - 1024)) * 64 + (mrow & 63)] = f2bf_bits(v);
            } else {
                Of[(size_t)mrow * CDIM + ncol] = v;
            }
        }
    }
}

// ---------- kernel 4: per-(window,head) attention, 2 waves ----------
__global__ __launch_bounds__(64)
void attn_wmma(const unsigned short* __restrict__ QK,   // [M_TOT][1024] bf16 (Q|K)
               const unsigned short* __restrict__ Vt,   // [WIN][512][64] bf16 (V transposed)
               unsigned short* __restrict__ ctx) {      // [M_TOT][512] bf16
    __shared__ __attribute__((aligned(16))) unsigned short Qs [64][72];
    __shared__ __attribute__((aligned(16))) unsigned short Ks [64][72];
    __shared__ __attribute__((aligned(16))) unsigned short VsT[64][72];  // [hd][key]
    __shared__ __attribute__((aligned(16))) unsigned short Ps [64][72];
    __shared__ __attribute__((aligned(16))) float          Ls [64][68];

    const int win  = blockIdx.x >> 3;
    const int head = blockIdx.x & 7;
    const int tid  = threadIdx.x;                 // 0..63
    const size_t qb = ((size_t)(win * S_TOK + tid)) * QK_LD + head * 64;
    const size_t vb = (((size_t)win) * CDIM + head * 64 + tid) * 64;   // Vt row: channel tid
    #pragma unroll
    for (int c8 = 0; c8 < 64; c8 += 8) {
        *(v8us*)&Qs [tid][c8] = *(const v8us*)&QK[qb + c8];
        *(v8us*)&Ks [tid][c8] = *(const v8us*)&QK[qb + 512 + c8];
        *(v8us*)&VsT[tid][c8] = *(const v8us*)&Vt[vb + c8];
    }
    __syncthreads();

    const int wv = tid >> 5, lane = tid & 31, lrow = lane & 15, lhi = lane >> 4;
    const float scale = 0.125f;                   // 1/sqrt(64)

    // logits = scale * Q @ K^T
    for (int it = 0; it < 2; ++it) {
        const int i = wv * 2 + it;
        #pragma unroll
        for (int j = 0; j < 4; ++j) {
            v8f acc = (v8f){0.f,0.f,0.f,0.f,0.f,0.f,0.f,0.f};
            #pragma unroll
            for (int kc = 0; kc < 64; kc += 32) {
                v16bf a  = frag_split16(&Qs[i * 16 + lrow][kc], lhi << 3);
                v16bf bb = frag_split16(&Ks[j * 16 + lrow][kc], lhi << 3);  // B[k][n]=K[n][k]
                acc = __builtin_amdgcn_wmma_f32_16x16x32_bf16(
                          false, a, false, bb, (short)0, acc, false, false);
            }
            #pragma unroll
            for (int r = 0; r < 8; ++r)
                Ls[i * 16 + (lhi << 3) + r][j * 16 + lrow] = acc[r] * scale;
        }
    }
    __syncthreads();

    // softmax: one row per thread, vectorized over float4
    {
        float4* lr = (float4*)&Ls[tid][0];
        float mx = -3.4e38f;
        #pragma unroll
        for (int c4 = 0; c4 < 16; ++c4) {
            float4 v = lr[c4];
            mx = fmaxf(mx, fmaxf(fmaxf(v.x, v.y), fmaxf(v.z, v.w)));
        }
        float sum = 0.f;
        #pragma unroll
        for (int c4 = 0; c4 < 16; ++c4) {
            float4 v = lr[c4];
            v.x = __expf(v.x - mx); v.y = __expf(v.y - mx);
            v.z = __expf(v.z - mx); v.w = __expf(v.w - mx);
            sum += v.x + v.y + v.z + v.w;
            lr[c4] = v;
        }
        float inv = 1.f / sum;
        #pragma unroll
        for (int c4 = 0; c4 < 16; ++c4) {
            float4 v = lr[c4];
            v4us pk;
            pk[0] = f2bf_bits(v.x * inv); pk[1] = f2bf_bits(v.y * inv);
            pk[2] = f2bf_bits(v.z * inv); pk[3] = f2bf_bits(v.w * inv);
            *(v4us*)&Ps[tid][c4 * 4] = pk;
        }
    }
    __syncthreads();

    // out = P @ V  (k-dim = 64 keys); B[k=key][n=hd] = VsT[hd][key], contiguous in key
    for (int it = 0; it < 2; ++it) {
        const int i = wv * 2 + it;
        #pragma unroll
        for (int j = 0; j < 4; ++j) {
            v8f acc = (v8f){0.f,0.f,0.f,0.f,0.f,0.f,0.f,0.f};
            #pragma unroll
            for (int kc = 0; kc < 64; kc += 32) {
                v16bf a  = frag_split16(&Ps[i * 16 + lrow][kc], lhi << 3);
                v16bf bb = frag_split16(&VsT[j * 16 + lrow][kc], lhi << 3);
                acc = __builtin_amdgcn_wmma_f32_16x16x32_bf16(
                          false, a, false, bb, (short)0, acc, false, false);
            }
            #pragma unroll
            for (int r = 0; r < 8; ++r) {
                int tok = win * 64 + i * 16 + (lhi << 3) + r;
                ctx[(size_t)tok * CDIM + head * 64 + j * 16 + lrow] = f2bf_bits(acc[r]);
            }
        }
    }
}

// ---------- kernel 6: un-window + crop + LayerNorm ----------
__global__ __launch_bounds__(128)
void unwindow_ln(const float* __restrict__ Yw,     // [M_TOT][512] f32 (windowed)
                 const float* __restrict__ lw,
                 const float* __restrict__ lb,
                 float* __restrict__ out) {        // [8][64][64][512] f32
    __shared__ float r1[128], r2[128];
    const int t   = blockIdx.x;
    const int b   = t >> 12;
    const int rem = t & 4095;
    const int h   = rem >> 6, w = rem & 63;
    const int hp  = h + 4, wp = w + 4;
    const int win = (b * 9 + (hp >> 3)) * 9 + (wp >> 3);
    const int s   = ((hp & 7) << 3) + (wp & 7);
    const float* __restrict__ row = Yw + ((size_t)win * 64 + s) * CDIM;
    const int tid = threadIdx.x;

    float4 v = *(const float4*)&row[tid * 4];
    float sum = v.x + v.y + v.z + v.w;
    float sq  = v.x * v.x + v.y * v.y + v.z * v.z + v.w * v.w;
    r1[tid] = sum; r2[tid] = sq;
    __syncthreads();
    for (int st = 64; st > 0; st >>= 1) {
        if (tid < st) { r1[tid] += r1[tid + st]; r2[tid] += r2[tid + st]; }
        __syncthreads();
    }
    float mu  = r1[0] * (1.f / 512.f);
    float var = r2[0] * (1.f / 512.f) - mu * mu;
    float inv = rsqrtf(var + 1e-5f);
    float4 g = *(const float4*)&lw[tid * 4];
    float4 o = *(const float4*)&lb[tid * 4];
    float4 y;
    y.x = (v.x - mu) * inv * g.x + o.x;
    y.y = (v.y - mu) * inv * g.y + o.y;
    y.z = (v.z - mu) * inv * g.z + o.z;
    y.w = (v.w - mu) * inv * g.w + o.w;
    *(float4*)&out[(size_t)t * CDIM + tid * 4] = y;
}

// ---------- launcher ----------
extern "C" void kernel_launch(void* const* d_in, const int* in_sizes, int n_in,
                              void* d_out, int out_size, void* d_ws, size_t ws_size,
                              hipStream_t stream) {
    const float* x    = (const float*)d_in[0];
    const float* res  = (const float*)d_in[1];
    const float* wqkv = (const float*)d_in[2];   // [1536][512]
    const float* bqkv = (const float*)d_in[3];   // [1536]
    const float* wout = (const float*)d_in[4];   // [512][512]
    const float* bout = (const float*)d_in[5];   // [512]
    const float* lnw  = (const float*)d_in[6];
    const float* lnb  = (const float*)d_in[7];
    float* out = (float*)d_out;

    char* ws = (char*)d_ws;
    size_t off = 0;
    unsigned short* Xq  = (unsigned short*)(ws + off); off += (size_t)M_TOT * CDIM * 2;      // 42.5 MB
    unsigned short* Xkv = (unsigned short*)(ws + off); off += (size_t)M_TOT * CDIM * 2;      // 42.5 MB
    unsigned short* Wqb = (unsigned short*)(ws + off); off += (size_t)3 * CDIM * CDIM * 2;   // 1.5 MB
    unsigned short* Wob = (unsigned short*)(ws + off); off += (size_t)CDIM * CDIM * 2;       // 0.5 MB
    unsigned short* QK  = (unsigned short*)(ws + off); off += (size_t)M_TOT * QK_LD * 2;     // 85 MB
    unsigned short* Vt  = (unsigned short*)(ws + off); off += (size_t)M_TOT * CDIM * 2;      // 42.5 MB
    unsigned short* Ctx = (unsigned short*)(ws + off); off += (size_t)M_TOT * CDIM * 2;      // 42.5 MB
    float*          Yw  = (float*)(ws + off);          off += (size_t)M_TOT * CDIM * 4;      // 85 MB

    const int total8 = WIN_TOT * S_TOK * CDIM / 8;
    window_bf16<<<(total8 + 255) / 256, 256, 0, stream>>>(x,   Xq);
    window_bf16<<<(total8 + 255) / 256, 256, 0, stream>>>(res, Xkv);
    cvt_bf16<<<(3 * CDIM * CDIM / 8 + 255) / 256, 256, 0, stream>>>(wqkv, Wqb, 3 * CDIM * CDIM / 8);
    cvt_bf16<<<(CDIM * CDIM / 8 + 255) / 256, 256, 0, stream>>>(wout, Wob, CDIM * CDIM / 8);

    // QKV projection: M=41472, N=1536, K=512
    gemm_wmma_bf16<<<dim3(M_TOT / 64, 24), 128, 0, stream>>>(
        Xq, Xkv, Wqb, bqkv, QK, Vt, nullptr, CDIM, 0);

    // attention per (window, head)
    attn_wmma<<<WIN_TOT * 8, 64, 0, stream>>>(QK, Vt, Ctx);

    // output projection: M=41472, N=512, K=512 (f32 out)
    gemm_wmma_bf16<<<dim3(M_TOT / 64, 8), 128, 0, stream>>>(
        Ctx, Ctx, Wob, bout, nullptr, nullptr, Yw, CDIM, 1);

    // un-window, crop 4-pixel border, LayerNorm
    unwindow_ln<<<8 * 64 * 64, 128, 0, stream>>>(Yw, lnw, lnb, out);
}
